// TCPGen_90348932039406
// MI455X (gfx1250) — matrix operand order repaired and checked by reading, compiled, not run on hardware
//
#include <hip/hip_runtime.h>
#include <hip/hip_bf16.h>
#include <math.h>

typedef __attribute__((ext_vector_type(16))) _Float16 v16h;
typedef __attribute__((ext_vector_type(8)))  _Float16 v8h;
typedef __attribute__((ext_vector_type(8)))  float    v8f;

#define BB   16
#define SS   200
#define TT   800
#define SR   5
#define VV   500
#define AD   512
#define MPB  (TT*SR)                 // 4000 rows per batch
#define MTOT (BB*MPB)                // 64000 rows total
#define DIST_OFF (MTOT*AD)           // 32,768,000
#define GEN_OFF  (DIST_OFF + MTOT*(VV+1))  // 64,832,000

union V16 { v16h v; v8h h[2]; };

// ---------------------------------------------------------------------------
// kv = concat(emb, ool) @ W_k + b_k  -> fp16 table (512x512, rows 501..511 = 0)
// and its transpose kvT[a][v] so both attention GEMMs get contiguous B-frags.
// ---------------------------------------------------------------------------
__global__ __launch_bounds__(256) void kv_build(
    const float* __restrict__ emb, const float* __restrict__ ool,
    const float* __restrict__ Wk,  const float* __restrict__ bk,
    _Float16* __restrict__ kv, _Float16* __restrict__ kvT) {
  __shared__ float xs[AD];
  const int v   = blockIdx.x;      // 0..511 (padded vocab)
  const int tid = threadIdx.x;     // 256 threads
  const bool valid = (v <= VV);
  const float* x = (v < VV) ? (emb + (size_t)v * AD) : ool;
  if (valid) { xs[tid] = x[tid]; xs[tid + 256] = x[tid + 256]; }
  __syncthreads();
  for (int rep = 0; rep < 2; ++rep) {
    const int a = tid + rep * 256;
    float acc = 0.0f;
    if (valid) {
      acc = bk[a];
      #pragma unroll 4
      for (int k = 0; k < AD; ++k) acc = fmaf(xs[k], Wk[k * AD + a], acc);
    }
    kv [(size_t)v * AD + a] = (_Float16)acc;
    kvT[(size_t)a * AD + v] = (_Float16)acc;
  }
}

// ---------------------------------------------------------------------------
// Row projections: rows [0, B*T) -> q_ac = enc @ W_qa + b_qa
//                  rows [B*T, B*T+B*S) -> q_sym = emb[targets] @ W_qs + b_qs
// ---------------------------------------------------------------------------
__global__ __launch_bounds__(256) void qproj(
    const float* __restrict__ enc, const float* __restrict__ emb,
    const int* __restrict__ targets,
    const float* __restrict__ Wqa, const float* __restrict__ bqa,
    const float* __restrict__ Wqs, const float* __restrict__ bqs,
    float* __restrict__ q_ac, float* __restrict__ q_sym) {
  __shared__ float xs[AD];
  const int row = blockIdx.x;
  const int tid = threadIdx.x;
  const float *x, *W, *bias; float* out;
  if (row < BB * TT) {
    x = enc + (size_t)row * AD; W = Wqa; bias = bqa; out = q_ac + (size_t)row * AD;
  } else {
    const int r2 = row - BB * TT;
    x = emb + (size_t)targets[r2] * AD; W = Wqs; bias = bqs; out = q_sym + (size_t)r2 * AD;
  }
  xs[tid] = x[tid]; xs[tid + 256] = x[tid + 256];
  __syncthreads();
  for (int rep = 0; rep < 2; ++rep) {
    const int a = tid + rep * 256;
    float acc = bias[a];
    #pragma unroll 4
    for (int k = 0; k < AD; ++k) acc = fmaf(xs[k], W[k * AD + a], acc);
    out[a] = acc;
  }
}

// ---------------------------------------------------------------------------
// Fused attention: per 16-row M-tile,
//   logits = (q_sym_p + q_ac) @ kv^T * scale   (WMMA f16, f32 accum)
//   dist   = masked softmax(logits)            (LDS)
//   hptr   = dist[:, :500] @ kv                (WMMA f16, f32 accum)
// 128 threads = 4 waves; each wave owns 8 of the 32 N-tiles per GEMM.
// ---------------------------------------------------------------------------
__global__ __launch_bounds__(128) void attn(
    const float* __restrict__ q_sym, const float* __restrict__ q_ac,
    const _Float16* __restrict__ kv, const _Float16* __restrict__ kvT,
    const int* __restrict__ ranges, const int* __restrict__ dist_masks,
    const int* __restrict__ gen_masks, float* __restrict__ out) {
  __shared__ _Float16 Qs[16 * 544];   // GEMM1 A-tile; reused as P-tile for GEMM2
  __shared__ float    Ls[16 * 520];   // logits / exp workspace (+8 reduction slots)
  __shared__ float    rowMax[16], rowSum[16];

  const int tid  = threadIdx.x;
  const int lane = tid & 31;
  const int wave = tid >> 5;
  const int hi   = lane >> 4;        // half-wave select
  const int lm   = lane & 15;
  const int Rbase = blockIdx.x * 16;

  // per-thread (row, segment) mapping for gather/softmax: 8 threads x 64 cols per row
  const int brow = tid >> 3;
  const int bseg = tid & 7;
  const int c0   = bseg * 64;
  const int R    = Rbase + brow;
  const int b    = R / MPB;
  const int rem  = R - b * MPB;
  const int t    = rem / SR;
  const int s    = rem - t * SR;
  const int srow = ranges[(b * TT + t) * SR + s];
  const int* mrow = dist_masks + (size_t)(b * SS + srow) * (VV + 1);

  // keep the L2-resident kv tables warm
  __builtin_prefetch(kv  + (size_t)tid * 512, 0, 3);
  __builtin_prefetch(kvT + (size_t)tid * 512, 0, 3);

  // ---- build 16x512 f16 query tile: gather q_sym rows + broadcast q_ac ----
  {
    const float* qs = q_sym + (size_t)(b * SS + srow) * AD;
    const float* qa = q_ac  + (size_t)(b * TT + t) * AD;
    for (int j = 0; j < 64; j += 4) {
      const float4 a4 = *(const float4*)(qs + c0 + j);
      const float4 b4 = *(const float4*)(qa + c0 + j);
      _Float16* dst = &Qs[brow * 544 + c0 + j];
      dst[0] = (_Float16)(a4.x + b4.x);
      dst[1] = (_Float16)(a4.y + b4.y);
      dst[2] = (_Float16)(a4.z + b4.z);
      dst[3] = (_Float16)(a4.w + b4.w);
    }
  }
  __syncthreads();

  const v8f vzero = {};
  v8f acc[8];

  // ---- GEMM1: logits[16 x 512] = Q[16 x 512] x kv^T (B[k][n] = kv[n][k]) ----
  #pragma unroll
  for (int nt = 0; nt < 8; ++nt) acc[nt] = vzero;
  for (int k0 = 0; k0 < AD; k0 += 32) {
    V16 af;  // A layout: lanes 0-15 M=lane K=0..15 ; lanes 16-31 K=16..31
    af.h[0] = *(const v8h*)&Qs[lm * 544 + k0 + hi * 16];
    af.h[1] = *(const v8h*)&Qs[lm * 544 + k0 + hi * 16 + 8];
    #pragma unroll
    for (int nt = 0; nt < 8; ++nt) {
      const int n0 = (wave * 8 + nt) * 16;
      const v16h bf = *(const v16h*)(kv + (size_t)(n0 + lm) * AD + k0 + hi * 16);
      acc[nt] = __builtin_amdgcn_wmma_f32_16x16x32_f16(
          false, af.v, false, bf, (short)0, acc[nt], false, false);
    }
  }
  const float SCALE = 0.04419417382415922f;  // 1/sqrt(512)
  #pragma unroll
  for (int nt = 0; nt < 8; ++nt) {
    const int n0 = (wave * 8 + nt) * 16;
    #pragma unroll
    for (int i = 0; i < 8; ++i)   // C/D layout: M = i + 8*hi, N = lm
      Ls[(hi * 8 + i) * 520 + n0 + lm] = acc[nt][i] * SCALE;
  }
  __syncthreads();

  // ---- masked softmax over 501 vocab entries (cols 501..511 are zero-padded) ----
  float m = -3.4028234663852886e38f;
  for (int j = 0; j < 64; ++j) {
    const int v = c0 + j;
    if (v < (VV + 1) && mrow[v] == 0) m = fmaxf(m, Ls[brow * 520 + v]);
  }
  Ls[brow * 520 + 512 + bseg] = m;
  __syncthreads();
  if (bseg == 0) {
    float mm = Ls[brow * 520 + 512];
    for (int q = 1; q < 8; ++q) mm = fmaxf(mm, Ls[brow * 520 + 512 + q]);
    rowMax[brow] = mm;
  }
  __syncthreads();
  const float rmax = rowMax[brow];
  float ssum = 0.0f;
  for (int j = 0; j < 64; ++j) {
    const int v = c0 + j;
    float e = 0.0f;
    if (v < (VV + 1) && mrow[v] == 0) e = __expf(Ls[brow * 520 + v] - rmax);
    Ls[brow * 520 + v] = e;
    ssum += e;
  }
  __syncthreads();
  Ls[brow * 520 + 512 + bseg] = ssum;
  __syncthreads();
  if (bseg == 0) {
    float sm = 0.0f;
    for (int q = 0; q < 8; ++q) sm += Ls[brow * 520 + 512 + q];
    rowSum[brow] = sm;
    out[GEN_OFF + R] = gen_masks[b * SS + srow] ? 1.0f : 0.0f;
  }
  __syncthreads();
  const float inv = 1.0f / rowSum[brow];
  float* distOut = out + DIST_OFF + (size_t)R * (VV + 1);
  for (int j = 0; j < 64; ++j) {
    const int v = c0 + j;
    const float p = Ls[brow * 520 + v] * inv;
    if (v < (VV + 1)) distOut[v] = p;
    Qs[brow * 544 + v] = (v < VV) ? (_Float16)p : (_Float16)0.0f;  // P-tile (OOL excluded)
  }
  __syncthreads();

  // ---- GEMM2: hptr[16 x 512] = P[16 x 512] x kv (B[k][n] = kvT[n][k]) ----
  #pragma unroll
  for (int nt = 0; nt < 8; ++nt) acc[nt] = vzero;
  for (int k0 = 0; k0 < AD; k0 += 32) {
    V16 af;
    af.h[0] = *(const v8h*)&Qs[lm * 544 + k0 + hi * 16];
    af.h[1] = *(const v8h*)&Qs[lm * 544 + k0 + hi * 16 + 8];
    #pragma unroll
    for (int nt = 0; nt < 8; ++nt) {
      const int n0 = (wave * 8 + nt) * 16;
      const v16h bf = *(const v16h*)(kvT + (size_t)(n0 + lm) * AD + k0 + hi * 16);
      acc[nt] = __builtin_amdgcn_wmma_f32_16x16x32_f16(
          false, af.v, false, bf, (short)0, acc[nt], false, false);
    }
  }
  #pragma unroll
  for (int nt = 0; nt < 8; ++nt) {
    const int n0 = (wave * 8 + nt) * 16;
    #pragma unroll
    for (int i = 0; i < 8; ++i)
      out[(size_t)(Rbase + hi * 8 + i) * AD + n0 + lm] = acc[nt][i];
  }
}

// ---------------------------------------------------------------------------
extern "C" void kernel_launch(void* const* d_in, const int* in_sizes, int n_in,
                              void* d_out, int out_size, void* d_ws, size_t ws_size,
                              hipStream_t stream) {
  const int*   targets = (const int*)  d_in[0];
  const float* enc     = (const float*)d_in[1];
  const int*   rngs    = (const int*)  d_in[2];
  const int*   distm   = (const int*)  d_in[3];
  const int*   genm    = (const int*)  d_in[4];
  const float* emb     = (const float*)d_in[5];
  const float* ool     = (const float*)d_in[6];
  const float* Wqs     = (const float*)d_in[7];
  const float* bqs     = (const float*)d_in[8];
  const float* Wqa     = (const float*)d_in[9];
  const float* bqa     = (const float*)d_in[10];
  const float* Wk      = (const float*)d_in[11];
  const float* bk      = (const float*)d_in[12];
  float* out = (float*)d_out;

  char* w = (char*)d_ws;
  _Float16* kv   = (_Float16*)(w);                         // 512*512*2  = 512 KB
  _Float16* kvT  = (_Float16*)(w + 524288);                // 512 KB
  float*    q_ac = (float*)   (w + 1048576);               // 12800*512*4 = 25 MB
  float*    q_sy = (float*)   (w + 1048576 + 26214400);    // 3200*512*4  = 6.25 MB

  kv_build<<<512, 256, 0, stream>>>(emb, ool, Wk, bk, kv, kvT);
  qproj<<<BB * TT + BB * SS, 256, 0, stream>>>(enc, emb, targets,
                                               Wqa, bqa, Wqs, bqs, q_ac, q_sy);
  attn<<<MTOT / 16, 128, 0, stream>>>(q_sy, q_ac, kv, kvT, rngs, distm, genm, out);
}
